// STAGCN_6399501271420
// MI455X (gfx1250) — matrix operand order repaired and checked by reading, not compiled, not run on hardware
//
#include <hip/hip_runtime.h>
#include <hip/hip_bf16.h>

// ---------------- problem constants ----------------
#define BB   32
#define F_IN 9
#define TT   256
#define VV   25
#define DH   128
#define NCLS 3
#define EE   128
#define TKK  5
#define EP   (EE + VV)        // 153 edges incl. self loops
#define EPSV 1e-5f
#define NSLP 0.2f
#define PTS  ((TT * VV) / 16) // 400 position tiles per batch item
#define NF   (16 * VV)        // 400 fc1 features
#define KFC  (DH * VV)        // 3200

typedef __attribute__((ext_vector_type(16))) __bf16 v16bf;
typedef __attribute__((ext_vector_type(8)))  __bf16 v8bf;
typedef __attribute__((ext_vector_type(8)))  float  v8f;
typedef __attribute__((ext_vector_type(4)))  unsigned int v4u;
typedef __attribute__((ext_vector_type(4)))  int    v4i;
typedef __attribute__((ext_vector_type(8)))  int    v8i;

__device__ inline v16bf cat8(v8bf lo, v8bf hi) {
  return __builtin_shufflevector(lo, hi, 0,1,2,3,4,5,6,7,8,9,10,11,12,13,14,15);
}
__device__ inline v8bf ld8(const __bf16* p) {           // 16B vector load
  return *reinterpret_cast<const v8bf*>(p);
}

// ---------------- Tensor Data Mover: 1D row copy global->LDS ----------------
#if __has_builtin(__builtin_amdgcn_tensor_load_to_lds)
#define HAVE_TDM 1
// Copy `elems` bf16 (contiguous) from gptr to LDS offset lds_off.
// D# per CDNA5 ISA: group0 = {count=1, lds_addr,
// global_addr[56:0], type=2}; group1 = {data_size=2B, tensor_dim0 =
// tile_dim0 = elems, tile_dim1 = 1, stride0 = elems}.
__device__ inline void tdm_load_row(unsigned lds_off, const void* gptr, int elems) {
  unsigned long long ga = (unsigned long long)(uintptr_t)gptr;
  v4u g0;
  g0.x = 1u;                                            // count=1, user mode
  g0.y = lds_off;                                       // LDS byte address
  g0.z = (unsigned)(ga & 0xFFFFFFFFull);                // global_addr[31:0]
  g0.w = (unsigned)((ga >> 32) & 0x01FFFFFFull) | (2u << 30); // [56:32] | type=2
  v8i g1;
  g1[0] = 0x00010000;            // workgroup_mask=0, data_size=1 (2 bytes)
  g1[1] = (elems & 0xFFFF) << 16;          // tensor_dim0[15:0] in bits 63:48
  g1[2] = (elems >> 16) | (1 << 16);       // tensor_dim0[31:16] | tensor_dim1=1
  g1[3] = (elems & 0xFFFF) << 16;          // tile_dim0 in bits 127:112
  g1[4] = 1;                               // tile_dim1=1, tile_dim2=0
  g1[5] = elems;                           // tensor_dim0_stride[31:0]
  g1[6] = 0;                               // stride0[47:32] | stride1[15:0]
  g1[7] = 0;                               // stride1[47:16]
  v4i z4 = {0, 0, 0, 0};
#if defined(__clang_major__) && __clang_major__ >= 23
  v8i z8 = {0, 0, 0, 0, 0, 0, 0, 0};
  __builtin_amdgcn_tensor_load_to_lds(g0, g1, z4, z4, z8, 0);
#else
  __builtin_amdgcn_tensor_load_to_lds(g0, g1, z4, z4, 0);
#endif
}
#endif

// =====================================================================
// WMMA temporal conv, bf16 activations.
// Workgroup = 8 waves = one 16-position tile x all 128 out channels.
// Input patch (K x 16) staged cooperatively in LDS (address affine in
// position: base_c + (kt-2)*V + pos), shared by all 8 waves; weights
// pre-packed [Dout][Kpad] so each lane's A fragment is two 16B loads.
// BN affine is folded into staging; T-padding stays zero.
// =====================================================================
#define KC   64            // K chunk per stage (2 WMMAs)
#define BSTR (KC + 8)      // LDS row stride (bank-decorrelated, 16B aligned)

__global__ __launch_bounds__(256)
void tconv_wmma_kernel(const __bf16* __restrict__ x,
                       const __bf16* __restrict__ wts,  // [DH][Kpad]
                       const float* __restrict__ bias,
                       const float* __restrict__ affine, // [2*Cin] or null
                       __bf16* __restrict__ out,
                       int Cin, int Kpad) {
  __shared__ __bf16 Bsg[16 * BSTR];
  const int wg   = blockIdx.x;            // b*PTS + pt
  const int pt   = wg % PTS, b = wg / PTS;
  const int tid  = threadIdx.x;
  const int wave = tid >> 5, lane = tid & 31;
  const int half = lane >> 4, idx = lane & 15;
  const int K    = Cin * TKK;
  const int pos0 = pt * 16;
  const int posl = pos0 + idx, tl = posl / VV, vl = posl % VV;
  const bool aff = (affine != nullptr);
  const __bf16* wrow = wts + (size_t)(wave * 16 + idx) * Kpad;  // A row m=idx

  v8f acc = {};
  for (int k0 = 0; k0 < K; k0 += KC) {
    __syncthreads();
    for (int e = tid; e < 16 * KC; e += 256) {   // stage B chunk, [n][k]
      const int kk = e >> 4, n = e & 15;
      const int kb = k0 + kk;
      float bv = 0.0f;
      if (kb < K) {
        const int c  = kb / TKK, kt = kb % TKK;
        const int p  = pos0 + n;
        const int t  = p / VV, v = p - t * VV;
        const int tt = t + kt - (TKK / 2);
        if (tt >= 0 && tt < TT) {
          bv = (float)x[((size_t)(b * Cin + c) * TT + tt) * VV + v];
          if (aff) bv = bv * affine[c] + affine[Cin + c];
        }
      }
      Bsg[n * BSTR + kk] = (__bf16)bv;
    }
    __syncthreads();
#pragma unroll
    for (int h2 = 0; h2 < 2; ++h2) {
      const int kc = k0 + h2 * 32;
      v16bf Af = cat8(ld8(wrow + kc + half * 8),
                      ld8(wrow + kc + 16 + half * 8));
      const __bf16* brow = &Bsg[idx * BSTR + h2 * 32 + half * 8];
      v16bf Bf = cat8(ld8(brow), ld8(brow + 16));
      acc = __builtin_amdgcn_wmma_f32_16x16x32_bf16(false, Af, false, Bf,
                                                    (short)0, acc, false, false);
    }
    if (k0 + KC < K)  // prefetch next weight chunk -> global_prefetch_b8
      __builtin_prefetch(wrow + k0 + KC + half * 8, 0, 3);
  }
#pragma unroll
  for (int r = 0; r < 8; ++r) {
    const int d = wave * 16 + r + half * 8;
    float o = acc[r] + bias[d];
    out[((size_t)(b * DH + d) * TT + tl) * VV + vl] = (__bf16)fmaxf(o, 0.0f);
  }
}

// =====================================================================
// WMMA node transform: y[(g*V+v), n] = sum_c h[b,c,t,v] * W[c,n].
// Workgroup = one 16-row tile x all 8 col tiles (wave = col tile).
// =====================================================================
#define ASTR (DH + 8)
__global__ __launch_bounds__(256)
void lin_wmma_kernel(const __bf16* __restrict__ h,
                     const __bf16* __restrict__ Wt,   // [DH][DH] transposed
                     __bf16* __restrict__ y) {
  __shared__ __bf16 Asg[16 * ASTR];
  const int rt   = blockIdx.x;                 // 12800 row tiles
  const int tid  = threadIdx.x;
  const int wave = tid >> 5, lane = tid & 31;
  const int half = lane >> 4, idx = lane & 15;
  const int rowBase = rt * 16;
  const int bA   = rowBase / (TT * VV);
  const int remB = rowBase % (TT * VV);        // 16 rows stay within one b

  for (int e = tid; e < 16 * DH; e += 256) {   // stage A [m][k], coalesced on m
    const int k = e >> 4, m = e & 15;
    Asg[m * ASTR + k] =
        h[(size_t)bA * DH * TT * VV + (size_t)k * TT * VV + remB + m];
  }
  __syncthreads();

  const __bf16* wrow = Wt + (size_t)(wave * 16 + idx) * DH;  // col n = idx
  v8f acc = {};
#pragma unroll
  for (int k0 = 0; k0 < DH; k0 += 32) {
    v16bf Af = cat8(ld8(&Asg[idx * ASTR + k0 + half * 8]),
                    ld8(&Asg[idx * ASTR + k0 + 16 + half * 8]));
    v16bf Bf = cat8(ld8(wrow + k0 + half * 8),
                    ld8(wrow + k0 + 16 + half * 8));
    acc = __builtin_amdgcn_wmma_f32_16x16x32_bf16(false, Af, false, Bf,
                                                  (short)0, acc, false, false);
  }
#pragma unroll
  for (int r = 0; r < 8; ++r) {
    const int row = rowBase + r + half * 8;
    y[(size_t)row * DH + wave * 16 + idx] = (__bf16)acc[r];
  }
}

// =====================================================================
// GATv2 attention + softmax + aggregation, one workgroup per frame-graph.
// xl/xr rows pulled into LDS by the Tensor Data Mover (wave 0 issues,
// waits TENSORcnt, barrier publishes); falls back to vector loads when
// the TDM builtin is unavailable. Accumulates per-channel BN stats.
// =====================================================================
__global__ __launch_bounds__(256)
void gat_kernel(const __bf16* __restrict__ xl, const __bf16* __restrict__ xr,
                const int* __restrict__ ei,
                const float* __restrict__ att, const float* __restrict__ gb,
                __bf16* __restrict__ out, float* __restrict__ stats) {
  __shared__ __bf16 xls[VV * DH];
  __shared__ __bf16 xrs[VV * DH];
  __shared__ float  outs[VV * DH];
  __shared__ float  es[EP], as[EP];
  __shared__ int    srcs[EP], dsts[EP];
  __shared__ float  segmax[VV], segsum[VV];

  const int g   = blockIdx.x;              // g = b*T + t
  const int tid = threadIdx.x;
  const size_t base = (size_t)g * VV * DH;

#if defined(HAVE_TDM)
  if (tid < 32) {                          // wave 0 launches both DMAs
    tdm_load_row((unsigned)(uintptr_t)&xls[0], xl + base, VV * DH);
    tdm_load_row((unsigned)(uintptr_t)&xrs[0], xr + base, VV * DH);
  }
#else
  for (int i = tid; i < VV * DH; i += 256) {
    xls[i] = xl[base + i];
    xrs[i] = xr[base + i];
  }
#endif
  for (int i = tid; i < VV * DH; i += 256) outs[i] = 0.0f;
  for (int j = tid; j < EP; j += 256) {
    srcs[j] = (j < EE) ? ei[j]      : (j - EE);
    dsts[j] = (j < EE) ? ei[EE + j] : (j - EE);
  }
#if defined(HAVE_TDM)
  if (tid < 32) __builtin_amdgcn_s_wait_tensorcnt(0);
#endif
  __syncthreads();

  const int wave = tid >> 5, lane = tid & 31;
  for (int j = wave; j < EP; j += 8) {
    const int s = srcs[j], d = dsts[j];
    float a = 0.0f;
#pragma unroll
    for (int c = lane; c < DH; c += 32) {
      float z = (float)xls[s * DH + c] + (float)xrs[d * DH + c];
      z = (z > 0.0f) ? z : NSLP * z;
      a += att[c] * z;
    }
#pragma unroll
    for (int o = 16; o > 0; o >>= 1) a += __shfl_xor(a, o, 32);
    if (lane == 0) es[j] = a;
  }
  __syncthreads();

  if (tid < VV) {
    float m = -1e30f;
    for (int j = 0; j < EP; ++j) if (dsts[j] == tid) m = fmaxf(m, es[j]);
    segmax[tid] = m;
    float s = 0.0f;
    for (int j = 0; j < EP; ++j) if (dsts[j] == tid) s += __expf(es[j] - m);
    segsum[tid] = s;
  }
  __syncthreads();
  for (int j = tid; j < EP; j += 256)
    as[j] = __expf(es[j] - segmax[dsts[j]]) / segsum[dsts[j]];
  __syncthreads();

  if (tid < DH) {
    const int c = tid;
    for (int j = 0; j < EP; ++j)
      outs[dsts[j] * DH + c] += as[j] * (float)xls[srcs[j] * DH + c];
  }
  __syncthreads();

  const int b = g / TT, t = g % TT;
  if (tid < DH) {
    const int c = tid;
    float lsum = 0.0f, lsq = 0.0f;
    for (int v = 0; v < VV; ++v) {
      float val = fmaxf(outs[v * DH + c] + gb[c], 0.0f);
      out[((size_t)(b * DH + c) * TT + t) * VV + v] = (__bf16)val;
      lsum += val; lsq += val * val;
    }
    atomicAdd(&stats[c],      lsum);
    atomicAdd(&stats[DH + c], lsq);
  }
}

// ---------------- BN stats -> folded (scale, shift) ----------------
__global__ void bnstats_kernel(const float* __restrict__ stats,
                               const float* __restrict__ gamma,
                               const float* __restrict__ beta,
                               float* __restrict__ ss) {
  int c = threadIdx.x;
  if (c < DH) {
    const float n  = (float)(BB * TT * VV);
    float mu  = stats[c] / n;
    float var = stats[DH + c] / n - mu * mu;
    float sc  = gamma[c] * rsqrtf(var + EPSV);
    ss[c]      = sc;
    ss[DH + c] = beta[c] - mu * sc;
  }
}

__global__ void zero_kernel(float* __restrict__ p, int n) {
  int i = blockIdx.x * blockDim.x + threadIdx.x;
  if (i < n) p[i] = 0.0f;
}

// ---------------- layout/precision prep kernels ----------------
__global__ void f32_to_bf16_kernel(const float* __restrict__ in,
                                   __bf16* __restrict__ out, int n) {
  int i = blockIdx.x * blockDim.x + threadIdx.x;
  if (i < n) out[i] = (__bf16)in[i];
}
__global__ void convw_kernel(const float* __restrict__ w,
                             __bf16* __restrict__ wts, int K, int Kpad, int Dout) {
  int i = blockIdx.x * blockDim.x + threadIdx.x;
  if (i < Dout * Kpad) {
    int d = i / Kpad, k = i % Kpad;
    wts[i] = (__bf16)((k < K) ? w[d * K + k] : 0.0f);
  }
}
__global__ void transw_kernel(const float* __restrict__ W,
                              __bf16* __restrict__ Wt, int Kdim, int Ndim) {
  int i = blockIdx.x * blockDim.x + threadIdx.x;
  if (i < Kdim * Ndim) {
    int n = i / Kdim, k = i % Kdim;
    Wt[i] = (__bf16)W[(size_t)k * Ndim + n];
  }
}

// ---------------- residual + temporal mean -> feat[B, DH*V] (bf16) --------
__global__ __launch_bounds__(256)
void residual_mean_kernel(const __bf16* __restrict__ x1,
                          const __bf16* __restrict__ x3,
                          __bf16* __restrict__ feat) {
  int i = blockIdx.x * blockDim.x + threadIdx.x;
  if (i >= BB * DH * VV) return;
  int b = i / (DH * VV), rem = i % (DH * VV);
  int d = rem / VV, v = rem % VV;
  size_t off = ((size_t)(b * DH + d) * TT) * VV + v;
  float s = 0.0f;
  for (int t = 0; t < TT; ++t)
    s += (float)x1[off + (size_t)t * VV] + (float)x3[off + (size_t)t * VV];
  feat[b * (DH * VV) + d * VV + v] = (__bf16)(s * (1.0f / TT));
}

// ---------------- FC1 via WMMA: h[32,400] = relu(feat @ W + b) -------------
__global__ __launch_bounds__(256)
void fc1_wmma_kernel(const __bf16* __restrict__ feat,   // [32][3200]
                     const __bf16* __restrict__ Wt,     // [400][3200]
                     const float* __restrict__ bias,
                     float* __restrict__ hout) {
  const int wid = blockIdx.x * (blockDim.x >> 5) + (threadIdx.x >> 5);
  if (wid >= 2 * 25) return;                 // whole-wave return
  const int ct = wid % 25, rt = wid / 25;
  const int lane = threadIdx.x & 31, half = lane >> 4, idx = lane & 15;
  const __bf16* arow = feat + (size_t)(rt * 16 + idx) * KFC;
  const __bf16* brow = Wt   + (size_t)(ct * 16 + idx) * KFC;

  v8f acc = {};
#pragma unroll 2
  for (int k0 = 0; k0 < KFC; k0 += 32) {
    v16bf Af = cat8(ld8(arow + k0 + half * 8), ld8(arow + k0 + 16 + half * 8));
    v16bf Bf = cat8(ld8(brow + k0 + half * 8), ld8(brow + k0 + 16 + half * 8));
    acc = __builtin_amdgcn_wmma_f32_16x16x32_bf16(false, Af, false, Bf,
                                                  (short)0, acc, false, false);
    __builtin_prefetch(brow + k0 + 64, 0, 3);
  }
#pragma unroll
  for (int r = 0; r < 8; ++r) {
    const int row = rt * 16 + r + half * 8;
    const int col = ct * 16 + idx;
    hout[row * NF + col] = fmaxf(acc[r] + bias[col], 0.0f);
  }
}

// ---------------- head: batch-norm over B, FC2, sigmoid ----------------
__global__ __launch_bounds__(256)
void head_kernel(const float* __restrict__ h,   // [32,400]
                 const float* __restrict__ bn_g, const float* __restrict__ bn_b,
                 const float* __restrict__ fc2_w, const float* __restrict__ fc2_b,
                 float* __restrict__ out) {
  __shared__ float hn[BB * NF];
  const int tid = threadIdx.x;
  for (int col = tid; col < NF; col += 256) {
    float mu = 0.0f, sq = 0.0f;
    for (int b = 0; b < BB; ++b) { float v = h[b * NF + col]; mu += v; sq += v * v; }
    mu *= (1.0f / BB);
    float var = sq * (1.0f / BB) - mu * mu;
    float sc  = bn_g[col] * rsqrtf(var + EPSV);
    float sh  = bn_b[col] - mu * sc;
    for (int b = 0; b < BB; ++b) hn[b * NF + col] = h[b * NF + col] * sc + sh;
  }
  __syncthreads();
  if (tid < BB * NCLS) {
    int b = tid / NCLS, cls = tid % NCLS;
    float l = fc2_b[cls];
    for (int j = 0; j < NF; ++j) l += hn[b * NF + j] * fc2_w[j * NCLS + cls];
    out[b * NCLS + cls] = 1.0f / (1.0f + __expf(-l));
  }
}

// =====================================================================
// host side
// =====================================================================
struct BlockW {
  const __bf16 *wt1; int kpad1;
  const float  *bt1;
  const __bf16 *wlt, *wrt;
  const float  *att, *gb, *gamma, *beta;
  const __bf16 *wt2; const float *bt2;
};

static void run_block(const __bf16* xin, int Cin, const BlockW& p, const int* ei,
                      __bf16* H, __bf16* XL, __bf16* XR,
                      float* stats, float* ss, __bf16* xout, hipStream_t stream) {
  zero_kernel<<<1, 256, 0, stream>>>(stats, 2 * DH);
  tconv_wmma_kernel<<<BB * PTS, 256, 0, stream>>>(xin, p.wt1, p.bt1, nullptr, H,
                                                  Cin, p.kpad1);
  lin_wmma_kernel<<<(BB * TT * VV) / 16, 256, 0, stream>>>(H, p.wlt, XL);
  lin_wmma_kernel<<<(BB * TT * VV) / 16, 256, 0, stream>>>(H, p.wrt, XR);
  gat_kernel<<<BB * TT, 256, 0, stream>>>(XL, XR, ei, p.att, p.gb, H, stats);
  bnstats_kernel<<<1, 128, 0, stream>>>(stats, p.gamma, p.beta, ss);
  tconv_wmma_kernel<<<BB * PTS, 256, 0, stream>>>(H, p.wt2, p.bt2, ss, xout,
                                                  DH, DH * TKK);
}

template <typename Tp>
static Tp* take(char*& p, size_t elems) {
  Tp* r = (Tp*)p;
  p += (elems * sizeof(Tp) + 255) & ~(size_t)255;
  return r;
}

extern "C" void kernel_launch(void* const* d_in, const int* in_sizes, int n_in,
                              void* d_out, int out_size, void* d_ws, size_t ws_size,
                              hipStream_t stream) {
  const float* x  = (const float*)d_in[0];
  const int*   ei = (const int*)d_in[1];
  // d_in[2] = train (ignored; eval mode)
  const float* p1f[10]; const float* p2f[10];
  for (int i = 0; i < 10; ++i) { p1f[i] = (const float*)d_in[3 + i];
                                 p2f[i] = (const float*)d_in[13 + i]; }
  const float* fc1_w = (const float*)d_in[23];
  const float* fc1_b = (const float*)d_in[24];
  const float* bn_g  = (const float*)d_in[25];
  const float* bn_b  = (const float*)d_in[26];
  const float* fc2_w = (const float*)d_in[27];
  const float* fc2_b = (const float*)d_in[28];

  const size_t ACT = (size_t)BB * DH * TT * VV;   // 26,214,400 elements
  const int K1 = F_IN * TKK, K1P = 64;            // 45 -> pad 64
  const int K2 = DH * TKK;                        // 640 (already /32)

  char* p = (char*)d_ws;
  __bf16* bufA  = take<__bf16>(p, ACT);   // x1 (kept for residual)
  __bf16* bufB  = take<__bf16>(p, ACT);   // x2
  __bf16* bufH  = take<__bf16>(p, ACT);   // tconv1 out / gat out
  __bf16* bufXL = take<__bf16>(p, ACT);
  __bf16* bufXR = take<__bf16>(p, ACT);   // also receives x3
  __bf16* xbf   = take<__bf16>(p, (size_t)BB * F_IN * TT * VV);
  __bf16* w1t1  = take<__bf16>(p, (size_t)DH * K1P);
  __bf16* w1t2  = take<__bf16>(p, (size_t)DH * K2);
  __bf16* w1l   = take<__bf16>(p, (size_t)DH * DH);
  __bf16* w1r   = take<__bf16>(p, (size_t)DH * DH);
  __bf16* w2t1  = take<__bf16>(p, (size_t)DH * K2);
  __bf16* w2t2  = take<__bf16>(p, (size_t)DH * K2);
  __bf16* w2l   = take<__bf16>(p, (size_t)DH * DH);
  __bf16* w2r   = take<__bf16>(p, (size_t)DH * DH);
  __bf16* fc1wt = take<__bf16>(p, (size_t)NF * KFC);
  __bf16* feat  = take<__bf16>(p, (size_t)BB * KFC);
  float*  hfc   = take<float>(p, (size_t)BB * NF);
  float*  stats = take<float>(p, 2 * DH);
  float*  ss    = take<float>(p, 2 * DH);
  if ((size_t)(p - (char*)d_ws) > ws_size) return;

  auto cdiv = [](size_t a, size_t b) { return (int)((a + b - 1) / b); };

  // ---- precision / layout prep ----
  const size_t NX = (size_t)BB * F_IN * TT * VV;
  f32_to_bf16_kernel<<<cdiv(NX, 256), 256, 0, stream>>>(x, xbf, (int)NX);
  convw_kernel<<<cdiv(DH * K1P, 256), 256, 0, stream>>>(p1f[0], w1t1, K1, K1P, DH);
  convw_kernel<<<cdiv(DH * K2, 256), 256, 0, stream>>>(p1f[8], w1t2, K2, K2, DH);
  convw_kernel<<<cdiv(DH * K2, 256), 256, 0, stream>>>(p2f[0], w2t1, K2, K2, DH);
  convw_kernel<<<cdiv(DH * K2, 256), 256, 0, stream>>>(p2f[8], w2t2, K2, K2, DH);
  transw_kernel<<<cdiv(DH * DH, 256), 256, 0, stream>>>(p1f[2], w1l, DH, DH);
  transw_kernel<<<cdiv(DH * DH, 256), 256, 0, stream>>>(p1f[3], w1r, DH, DH);
  transw_kernel<<<cdiv(DH * DH, 256), 256, 0, stream>>>(p2f[2], w2l, DH, DH);
  transw_kernel<<<cdiv(DH * DH, 256), 256, 0, stream>>>(p2f[3], w2r, DH, DH);
  transw_kernel<<<cdiv((size_t)KFC * NF, 256), 256, 0, stream>>>(fc1_w, fc1wt, KFC, NF);

  BlockW b1 = { w1t1, K1P, p1f[1], w1l, w1r, p1f[4], p1f[5], p1f[6], p1f[7],
                w1t2, p1f[9] };
  BlockW b2 = { w2t1, K2,  p2f[1], w2l, w2r, p2f[4], p2f[5], p2f[6], p2f[7],
                w2t2, p2f[9] };

  run_block(xbf,  F_IN, b1, ei, bufH, bufXL, bufXR, stats, ss, bufA,  stream); // x1
  run_block(bufA, DH,   b2, ei, bufH, bufXL, bufXR, stats, ss, bufB,  stream); // x2
  run_block(bufB, DH,   b2, ei, bufH, bufXL, bufXR, stats, ss, bufXR, stream); // x3

  residual_mean_kernel<<<(BB * DH * VV) / 256, 256, 0, stream>>>(bufA, bufXR, feat);
  fc1_wmma_kernel<<<7, 256, 0, stream>>>(feat, fc1wt, fc1_b, hfc);
  head_kernel<<<1, 256, 0, stream>>>(hfc, bn_g, bn_b, fc2_w, fc2_b, (float*)d_out);
}
